// RNNDecoder_50173807952073
// MI455X (gfx1250) — matrix operand order repaired and checked by reading, not compiled, not run on hardware
//
#include <hip/hip_runtime.h>
#include <math.h>

typedef __attribute__((ext_vector_type(16))) __bf16 v16bf;
typedef __attribute__((ext_vector_type(8)))  __bf16 v8bf;
typedef __attribute__((ext_vector_type(8)))  float  v8f;

#define BATCH  4096
#define LAT    128
#define OBS    64
#define HID    1024
#define KC     (OBS + HID)     // 1088 combined K for [x|h] @ [W_ih|W_hh]^T
#define KT_C   (KC / 32)       // 34
#define NT_H   (HID / 16)      // 64
#define KT_H   (HID / 32)      // 32
#define NT_O   (OBS / 16)      // 4
#define TILE_B 32              // batch rows per workgroup (2 M-tiles)
#define ROWSTRIDE 1160         // bf16 elems; 580 dwords, 580 % 64 == 4 -> bank-friendly
#define NWAVES 16
#define BLOCK  (NWAVES * 32)   // 512 threads
#define FRAG   512             // bf16 elements per packed 32x16 B-fragment

// Native TRANS32 tanh on CDNA5 (probe-confirmed: valu count collapsed).
__device__ __forceinline__ float fast_tanhf(float x) {
#if __has_builtin(__builtin_amdgcn_tanhf)
    return __builtin_amdgcn_tanhf(x);
#else
    return tanhf(x);
#endif
}

// ---------------------------------------------------------------------------
// Weight packing: B-matrix (K x N = W^T) fragments in this code's contiguous-K bf16 B
// layout: lane L -> column N = nt*16 + (L&15), 16 consecutive K starting at
// kt*32 + 16*(L>>4).  Stored [frag][g][lane][8] -> coalesced b128 pairs.
// ---------------------------------------------------------------------------
__global__ void pack_wc_kernel(const float* __restrict__ W_ih,
                               const float* __restrict__ W_hh,
                               __bf16* __restrict__ Wc) {
    int t = blockIdx.x * 256 + threadIdx.x;          // KT_C*NT_H*64 threads
    if (t >= KT_C * NT_H * 64) return;
    int f  = t >> 6;
    int g  = (t >> 5) & 1;
    int L  = t & 31;
    int kt = f / NT_H, nt = f % NT_H;
    int n     = nt * 16 + (L & 15);
    int kbase = kt * 32 + 16 * (L >> 4) + g * 8;
    __bf16* dst = Wc + (size_t)f * FRAG + g * 256 + L * 8;
    #pragma unroll
    for (int e = 0; e < 8; ++e) {
        int K = kbase + e;
        float v = (K < OBS) ? W_ih[(size_t)n * OBS + K]
                            : W_hh[(size_t)n * HID + (K - OBS)];
        dst[e] = (__bf16)v;
    }
}

__global__ void pack_who_kernel(const float* __restrict__ W_ho,
                                __bf16* __restrict__ Who) {
    int t = blockIdx.x * 256 + threadIdx.x;          // KT_H*NT_O*64 threads
    if (t >= KT_H * NT_O * 64) return;
    int f  = t >> 6;
    int g  = (t >> 5) & 1;
    int L  = t & 31;
    int kt = f / NT_O, nt = f % NT_O;
    int n     = nt * 16 + (L & 15);
    int kbase = kt * 32 + 16 * (L >> 4) + g * 8;
    __bf16* dst = Who + (size_t)f * FRAG + g * 256 + L * 8;
    #pragma unroll
    for (int e = 0; e < 8; ++e)
        dst[e] = (__bf16)W_ho[(size_t)n * HID + (kbase + e)];
}

__global__ void pack_bias_kernel(const float* __restrict__ b_ih,
                                 const float* __restrict__ b_hh,
                                 float* __restrict__ bc) {
    int j = blockIdx.x * 256 + threadIdx.x;
    if (j < HID) bc[j] = b_ih[j] + b_hh[j];
}

// ---------------------------------------------------------------------------
// Fragment loaders
// ---------------------------------------------------------------------------
__device__ __forceinline__ v16bf load_afrag(const __bf16* rowPtr, int kOff, int half) {
    const v8bf lo = *(const v8bf*)(rowPtr + kOff + 16 * half);
    const v8bf hi = *(const v8bf*)(rowPtr + kOff + 16 * half + 8);
    v16bf a;
    #pragma unroll
    for (int i = 0; i < 8; ++i) { a[i] = lo[i]; a[i + 8] = hi[i]; }
    return a;
}

__device__ __forceinline__ v16bf load_bfrag(const __bf16* base, int lane) {
    const v8bf lo = *(const v8bf*)(base + lane * 8);
    const v8bf hi = *(const v8bf*)(base + 256 + lane * 8);
    v16bf b;
    #pragma unroll
    for (int i = 0; i < 8; ++i) { b[i] = lo[i]; b[i + 8] = hi[i]; }
    return b;
}

// ---------------------------------------------------------------------------
// Persistent per-batch-tile RNN kernel.  One block owns TILE_B batch rows for
// all T steps; activations [x(64) | h(1024)] live in LDS as bf16.
// Wave w owns a 2-mTile x 4-nTile register block (nTiles w*4..w*4+3): every
// B fragment is loaded exactly once per WG per step and reused across mTiles.
// ---------------------------------------------------------------------------
__global__ __launch_bounds__(BLOCK, 1)
void rnn_main_kernel(const float* __restrict__ z,
                     const float* __restrict__ W_l2h,
                     const float* __restrict__ b_l2h,
                     const __bf16* __restrict__ Wc,
                     const __bf16* __restrict__ Who,
                     const float* __restrict__ bc,
                     const float* __restrict__ b_ho,
                     const int* __restrict__ seq_len_p,
                     float* __restrict__ out) {
    __shared__ __bf16 sAct[TILE_B * ROWSTRIDE];   // 74,240 bytes (CDNA5: 320KB/WGP)

    const int T     = *seq_len_p;
    const int bBase = blockIdx.x * TILE_B;
    const int tid   = threadIdx.x;
    const int lane  = tid & 31;
    const int wave  = tid >> 5;      // 0..15 -> nTiles wave*4 .. wave*4+3
    const int half  = lane >> 4;
    const int mrow  = lane & 15;
    const int ncol  = lane & 15;

    // ---- Prologue: h0 = z @ W_l2h^T + b_l2h (fp32, one-time); x0 = 0 ------
    for (int idx = tid; idx < TILE_B * HID; idx += BLOCK) {
        int r = idx >> 10, c = idx & (HID - 1);
        const float* zr = z + (size_t)(bBase + r) * LAT;
        const float* wr = W_l2h + (size_t)c * LAT;
        float acc = b_l2h[c];
        #pragma unroll 8
        for (int k = 0; k < LAT; ++k) acc += zr[k] * wr[k];
        sAct[r * ROWSTRIDE + OBS + c] = (__bf16)acc;
    }
    for (int idx = tid; idx < TILE_B * OBS; idx += BLOCK) {
        int r = idx >> 6, c = idx & (OBS - 1);
        sAct[r * ROWSTRIDE + c] = (__bf16)0.0f;
    }
    __syncthreads();

    const __bf16* rowPtr0 = sAct + (size_t)mrow * ROWSTRIDE;          // mTile 0
    const __bf16* rowPtr1 = sAct + (size_t)(16 + mrow) * ROWSTRIDE;   // mTile 1

    for (int t = 0; t < T; ++t) {
        // ---- h' = tanh([x|h] @ [W_ih|W_hh]^T + (b_ih+b_hh)) ----------------
        v8f acc0[4], acc1[4];
        #pragma unroll
        for (int j = 0; j < 4; ++j)
            #pragma unroll
            for (int r = 0; r < 8; ++r) { acc0[j][r] = 0.0f; acc1[j][r] = 0.0f; }

        for (int kt = 0; kt < KT_C; ++kt) {
            v16bf a0 = load_afrag(rowPtr0, kt * 32, half);
            v16bf a1 = load_afrag(rowPtr1, kt * 32, half);
            const __bf16* fb = Wc + (size_t)(kt * NT_H + wave * 4) * FRAG;
            #pragma unroll
            for (int j = 0; j < 4; ++j) {
                v16bf b = load_bfrag(fb + (size_t)j * FRAG, lane);
                acc0[j] = __builtin_amdgcn_wmma_f32_16x16x32_bf16(
                    false, a0, false, b, (short)0, acc0[j], false, false);
                acc1[j] = __builtin_amdgcn_wmma_f32_16x16x32_bf16(
                    false, a1, false, b, (short)0, acc1[j], false, false);
            }
        }
        __syncthreads();   // everyone done reading old x|h

        #pragma unroll
        for (int j = 0; j < 4; ++j) {
            int n = (wave * 4 + j) * 16 + ncol;
            float bias = bc[n];
            #pragma unroll
            for (int r = 0; r < 8; ++r) {
                int rr = r + 8 * half;
                float v0 = fast_tanhf(acc0[j][r] + bias);
                float v1 = fast_tanhf(acc1[j][r] + bias);
                sAct[(size_t)rr * ROWSTRIDE + OBS + n]        = (__bf16)v0;
                sAct[(size_t)(16 + rr) * ROWSTRIDE + OBS + n] = (__bf16)v1;
            }
        }
        __syncthreads();   // h' visible

        // ---- y = h' @ W_ho^T + b_ho ; feed back as next x ------------------
        // 2 mTiles x 4 nTiles = 8 output tiles -> waves 0..7
        if (wave < 2 * NT_O) {
            const int ym = wave >> 2;          // 0..1
            const int yn = wave & 3;           // 0..3
            const __bf16* yRowPtr = sAct + (size_t)(ym * 16 + mrow) * ROWSTRIDE;
            v8f accy;
            #pragma unroll
            for (int r = 0; r < 8; ++r) accy[r] = 0.0f;
            for (int kt = 0; kt < KT_H; ++kt) {
                v16bf a = load_afrag(yRowPtr, OBS + kt * 32, half);
                v16bf b = load_bfrag(Who + (size_t)(kt * NT_O + yn) * FRAG, lane);
                accy = __builtin_amdgcn_wmma_f32_16x16x32_bf16(
                    false, a, false, b, (short)0, accy, false, false);
            }
            int n = yn * 16 + ncol;
            float bias = b_ho[n];
            #pragma unroll
            for (int r = 0; r < 8; ++r) {
                int row = ym * 16 + r + 8 * half;
                float v = accy[r] + bias;
                out[((size_t)(bBase + row) * T + t) * OBS + n] = v;
                sAct[(size_t)row * ROWSTRIDE + n] = (__bf16)v;   // x_{t+1}
            }
        }
        __syncthreads();   // x ready for next step
    }
}

// ---------------------------------------------------------------------------
extern "C" void kernel_launch(void* const* d_in, const int* in_sizes, int n_in,
                              void* d_out, int out_size, void* d_ws, size_t ws_size,
                              hipStream_t stream) {
    const float* z      = (const float*)d_in[0];
    const float* W_l2h  = (const float*)d_in[1];
    const float* b_l2h  = (const float*)d_in[2];
    const float* W_ih   = (const float*)d_in[3];
    const float* b_ih   = (const float*)d_in[4];
    const float* W_hh   = (const float*)d_in[5];
    const float* b_hh   = (const float*)d_in[6];
    const float* W_ho   = (const float*)d_in[7];
    const float* b_ho   = (const float*)d_in[8];
    const int*   seqp   = (const int*)d_in[9];
    float*       out    = (float*)d_out;

    char* ws = (char*)d_ws;
    const size_t wcBytes  = (size_t)KT_C * NT_H * FRAG * sizeof(__bf16);   // ~2.2 MB
    const size_t whoBytes = (size_t)KT_H * NT_O * FRAG * sizeof(__bf16);   // 128 KB
    __bf16* Wc  = (__bf16*)ws;
    __bf16* Who = (__bf16*)(ws + wcBytes);
    float*  bc  = (float*)(ws + wcBytes + whoBytes);

    pack_wc_kernel <<<(KT_C * NT_H * 64 + 255) / 256, 256, 0, stream>>>(W_ih, W_hh, Wc);
    pack_who_kernel<<<(KT_H * NT_O * 64 + 255) / 256, 256, 0, stream>>>(W_ho, Who);
    pack_bias_kernel<<<(HID + 255) / 256, 256, 0, stream>>>(b_ih, b_hh, bc);

    rnn_main_kernel<<<BATCH / TILE_B, BLOCK, 0, stream>>>(
        z, W_l2h, b_l2h, Wc, Who, bc, b_ho, seqp, out);
}
